// Cells_46523085750313
// MI455X (gfx1250) — matrix operand order repaired and checked by reading, not compiled, not run on hardware
//
#include <hip/hip_runtime.h>

typedef __attribute__((ext_vector_type(16))) _Float16 v16h;
typedef __attribute__((ext_vector_type(8)))  float    v8f;

__device__ __forceinline__ v8f wmma_f16(v16h a, v16h b, v8f c) {
  // D = A(16x32,f16) x B(32x16,f16) + C(16x16,f32)
  return __builtin_amdgcn_wmma_f32_16x16x32_f16(false, a, false, b, (short)0, c, false, false);
}

// ---------------- problem dims ----------------
constexpr int B_ = 256, N_ = 35, I_ = 64, H_ = 256, R_ = 12;
constexpr int NN  = N_ * N_;        // 1225
constexpr int NI  = N_ * I_;        // 2240
constexpr int NH  = N_ * H_;        // 8960
constexpr size_t OUT1 = (size_t)B_ * NH;   // 2,293,760 floats per GRU output

// ---------------- workspace layout (bytes) ----------------
constexpr size_t SZ_A    = (size_t)R_ * NN * sizeof(float);                  // 58,800
constexpr size_t OFF_A1  = 0;
constexpr size_t OFF_A2  = (OFF_A1 + SZ_A + 255) & ~(size_t)255;
constexpr size_t SZ_W    = (size_t)R_ * 2 * 16 * 32 * 16 * sizeof(unsigned short); // 393,216
constexpr size_t OFF_W1  = (OFF_A2 + SZ_A + 255) & ~(size_t)255;
constexpr size_t OFF_W2  = OFF_W1 + SZ_W;
constexpr size_t SZ_G    = (size_t)7 * 16 * 16 * 32 * 16 * sizeof(unsigned short); // 1,835,008
constexpr size_t OFF_WR  = OFF_W2 + SZ_W;
constexpr size_t OFF_WZ  = OFF_WR + SZ_G;
constexpr size_t OFF_WH  = OFF_WZ + SZ_G;
constexpr size_t SZ_O    = (size_t)R_ * B_ * NH * sizeof(float);             // 110,100,480
constexpr size_t OFF_O   = OFF_WH + SZ_G;
constexpr size_t SZ_COM  = (size_t)4 * B_ * NH * sizeof(float);              // 36,700,160
constexpr size_t OFF_COM = OFF_O + SZ_O;
constexpr size_t OFF_SIM = OFF_COM + SZ_COM;                                 // 6*256 floats

struct P12 { const float* p[12]; };
struct P7  { const float* p[7];  };

// =====================================================================
// K0a: 12 relational adjacency compositions for A and for A^2 (35x35)
// =====================================================================
__global__ __launch_bounds__(256) void compose_kernel(
    const float* __restrict__ As, const float* __restrict__ Af,
    const float* __restrict__ At, float* __restrict__ A1, float* __restrict__ A2) {
  __shared__ float Ms[3][NN];
  __shared__ float Mp[6][NN];
  for (int s = 0; s < 2; ++s) {
    float* dst = (s == 0) ? A1 : A2;
    for (int e = threadIdx.x; e < NN; e += 256) {
      float a = As[e], f = Af[e], t = At[e];
      if (s == 1) { a *= a; f *= f; t *= t; }
      Ms[0][e] = a; Ms[1][e] = f; Ms[2][e] = t;
      dst[0 * NN + e] = a; dst[1 * NN + e] = f; dst[2 * NN + e] = t;
    }
    __syncthreads();
    const int pl[6] = {0, 1, 0, 2, 1, 2};
    const int pr[6] = {1, 0, 2, 0, 2, 1};     // As@Af, Af@As, As@At, At@As, Af@At, At@Af
    for (int p = 0; p < 6; ++p)
      for (int e = threadIdx.x; e < NN; e += 256) {
        int m = e / N_, n = e % N_;
        float acc = 0.f;
        for (int q = 0; q < N_; ++q) acc += Ms[pl[p]][m * N_ + q] * Ms[pr[p]][q * N_ + n];
        Mp[p][e] = acc; dst[(3 + p) * NN + e] = acc;
      }
    __syncthreads();
    const int tl[3] = {0, 2, 3};              // (As@Af)@At, (As@At)@Af, (At@As)@Af
    const int tr[3] = {2, 1, 1};
    for (int t = 0; t < 3; ++t)
      for (int e = threadIdx.x; e < NN; e += 256) {
        int m = e / N_, n = e % N_;
        float acc = 0.f;
        for (int q = 0; q < N_; ++q) acc += Mp[tl[t]][m * N_ + q] * Ms[tr[t]][q * N_ + n];
        dst[(9 + t) * NN + e] = acc;
      }
    __syncthreads();
  }
}

// =====================================================================
// K0b: W_r = sum_b comp[r,b] V[b]  -> f16, WMMA B-fragment swizzle
// B layout (32x16 f16): lane = half*16 + n%16 ; elem i -> k = 16*(i>>3)+8*half+(i&7)
// index = (((r*2+kstep)*16+ntile)*32+lane)*16 + i
// =====================================================================
__global__ __launch_bounds__(256) void gcnw_kernel(
    const float* __restrict__ V1, const float* __restrict__ comp1,
    const float* __restrict__ V2, const float* __restrict__ comp2,
    _Float16* __restrict__ W1s, _Float16* __restrict__ W2s) {
  int idx = blockIdx.x * 256 + threadIdx.x;       // 2 * 196,608 total
  if (idx >= 2 * R_ * 2 * 16 * 32 * 16) return;
  int which = idx / (R_ * 16384);
  int rem   = idx % (R_ * 16384);
  int i = rem & 15, lane = (rem >> 4) & 31, ntile = (rem >> 9) & 15;
  int kstep = (rem >> 13) & 1, r = rem >> 14;
  int n = ntile * 16 + (lane & 15);
  int k = kstep * 32 + ((i >> 3) << 4) + ((lane >> 4) << 3) + (i & 7);
  const float* V = which ? V2 : V1;
  const float* c = which ? comp2 : comp1;
  float acc = 0.f;
  for (int bb = 0; bb < 4; ++bb) acc += c[r * 4 + bb] * V[((size_t)bb * I_ + k) * H_ + n];
  (which ? W2s : W1s)[rem] = (_Float16)acc;
}

// =====================================================================
// K0c: Wr/Wz/Wh [7][512][256] f32 -> f16 B-fragment swizzle
// index = (((g*16+kstep)*16+ntile)*32+lane)*16 + i  (k = kstep*32 + 16*(i>>3) + 8*half + (i&7))
// =====================================================================
__global__ __launch_bounds__(256) void gatew_kernel(
    const float* __restrict__ Wr, const float* __restrict__ Wz, const float* __restrict__ Wh,
    _Float16* __restrict__ WrS, _Float16* __restrict__ WzS, _Float16* __restrict__ WhS) {
  int idx = blockIdx.x * 256 + threadIdx.x;       // 3 * 917,504 total
  if (idx >= 3 * 7 * 131072) return;
  int gate = idx / (7 * 131072);
  int rem  = idx % (7 * 131072);
  int i = rem & 15, lane = (rem >> 4) & 31, ntile = (rem >> 9) & 15;
  int kstep = (rem >> 13) & 15, g = rem >> 17;
  int n = ntile * 16 + (lane & 15);
  int k = kstep * 32 + ((i >> 3) << 4) + ((lane >> 4) << 3) + (i & 7);
  const float* W = (gate == 0) ? Wr : (gate == 1) ? Wz : Wh;
  _Float16* D    = (gate == 0) ? WrS : (gate == 1) ? WzS : WhS;
  D[rem] = (_Float16)W[((size_t)g * 512 + k) * H_ + n];
}

// =====================================================================
// K1: GCN.  Per (r,b): Y = A*x (VALU) -> LDS f16 A-frag swizzle ->
// WMMA Y@W per base, relu, sum, store O[r][b][35][256].
// A layout (16x32 f16): lane = half*16 + m%16; elem i -> k = 16*(i>>3)+8*half+(i&7)
// =====================================================================
__global__ __launch_bounds__(256) void gcn_kernel(
    P12 X, const float* __restrict__ A1, const float* __restrict__ A2,
    const _Float16* __restrict__ W1s, const _Float16* __restrict__ W2s,
    float* __restrict__ O) {
  int r = blockIdx.y, b = blockIdx.x;
  __shared__ float xls[NI];
  __shared__ float a1s[NN], a2s[NN];
  __shared__ __align__(32) _Float16 Ysw[2 * 3 * 2 * 512];   // [gcn][mtile][kstep][lane][i]
  const float* x = X.p[r] + (size_t)b * NI;
  for (int e = threadIdx.x; e < NI; e += 256) xls[e] = x[e];
  for (int e = threadIdx.x; e < NN; e += 256) { a1s[e] = A1[r * NN + e]; a2s[e] = A2[r * NN + e]; }
  for (int e = threadIdx.x; e < 6144; e += 256) Ysw[e] = (_Float16)0.f;
  __syncthreads();
  for (int e = threadIdx.x; e < 2 * NI; e += 256) {
    int gcn = e / NI, rem = e % NI;
    int m = rem >> 6, k = rem & 63;
    const float* A = gcn ? a2s : a1s;
    float acc = 0.f;
    for (int q = 0; q < N_; ++q) acc += A[m * N_ + q] * xls[q * I_ + k];
    int kstep = k >> 5, kk = k & 31;
    int hl = (kk >> 3) & 1;
    int i  = ((kk >> 4) << 3) | (kk & 7);
    int lane = hl * 16 + (m & 15);
    int mtile = m >> 4;
    Ysw[(((gcn * 3 + mtile) * 2 + kstep) << 9) + lane * 16 + i] = (_Float16)acc;
  }
  __syncthreads();
  int wave = threadIdx.x >> 5, lane = threadIdx.x & 31;
  int hl = lane >> 4;
  for (int t = 0; t < 6; ++t) {
    int tile = wave * 6 + t;                 // 48 tiles: mtile in [0,3), ntile in [0,16)
    int mtile = tile >> 4, ntile = tile & 15;
    v8f acc1 = {0.f,0.f,0.f,0.f,0.f,0.f,0.f,0.f};
    v8f acc2 = {0.f,0.f,0.f,0.f,0.f,0.f,0.f,0.f};
    for (int ks = 0; ks < 2; ++ks) {
      v16h a1v = *(const v16h*)&Ysw[((mtile * 2 + ks) << 9) + lane * 16];
      v16h b1v = *(const v16h*)&W1s[(((r * 2 + ks) * 16 + ntile) * 32 + lane) * 16];
      acc1 = wmma_f16(a1v, b1v, acc1);
      v16h a2v = *(const v16h*)&Ysw[(((3 + mtile) * 2 + ks) << 9) + lane * 16];
      v16h b2v = *(const v16h*)&W2s[(((r * 2 + ks) * 16 + ntile) * 32 + lane) * 16];
      acc2 = wmma_f16(a2v, b2v, acc2);
    }
    int n = ntile * 16 + (lane & 15);
    for (int j = 0; j < 8; ++j) {
      int m = mtile * 16 + j + hl * 8;       // D: VGPR j -> M=j (lanes 0-15), M=j+8 (16-31)
      if (m < N_) {
        float v = fmaxf(acc1[j], 0.f) + fmaxf(acc2[j], 0.f);
        O[(((size_t)r * B_ + b) * N_ + m) * H_ + n] = v;
      }
    }
  }
}

// =====================================================================
// K2: msh + consensus + pairwise MSE partials.  group g in {sf,st,ft,sft}
// m[b,mn,h] = sum_n o[b,n,h] * S[n,mn].  h chunked (2x128) to fit LDS.
// =====================================================================
__global__ __launch_bounds__(256) void msh_kernel(
    const float* __restrict__ O,
    const float* __restrict__ Ssf, const float* __restrict__ Sst,
    const float* __restrict__ Sft, const float* __restrict__ Ssft,
    float* __restrict__ com, float* __restrict__ simPart) {
  int g = blockIdx.y, b = blockIdx.x, tid = threadIdx.x;
  __shared__ float Ss[NN];
  __shared__ float M[3][N_ * 128];
  __shared__ float red[256];
  const float* S = (g == 0) ? Ssf : (g == 1) ? Sst : (g == 2) ? Sft : Ssft;
  int nm = (g == 3) ? 3 : 2;
  int r0 = (g == 3) ? 9 : 3 + 2 * g;
  for (int e = tid; e < NN; e += 256) Ss[e] = S[e];
  __syncthreads();
  float pacc[3] = {0.f, 0.f, 0.f};
  float invn = (nm == 3) ? (1.f / 3.f) : 0.5f;
  float* comp_ = com + ((size_t)g * B_ + b) * NH;
  for (int c = 0; c < 2; ++c) {
    for (int mem = 0; mem < nm; ++mem) {
      const float* Op = O + ((size_t)(r0 + mem) * B_ + b) * NH;
      for (int e = tid; e < N_ * 128; e += 256) {
        int mn = e >> 7, h = (c << 7) + (e & 127);
        float acc = 0.f;
        for (int q = 0; q < N_; ++q) acc += Op[q * H_ + h] * Ss[q * N_ + mn];
        M[mem][e] = acc;
      }
    }
    for (int e = tid; e < N_ * 128; e += 256) {   // each thread reads only its own entries
      int mn = e >> 7, hh = e & 127;
      float m0 = M[0][e], m1 = M[1][e], m2 = (nm == 3) ? M[2][e] : 0.f;
      comp_[mn * H_ + (c << 7) + hh] = (m0 + m1 + m2) * invn;
      float d0 = m0 - m1; pacc[0] += d0 * d0;
      if (nm == 3) { float d1 = m0 - m2; pacc[1] += d1 * d1;
                     float d2 = m1 - m2; pacc[2] += d2 * d2; }
    }
    __syncthreads();
  }
  int npairs = (nm == 3) ? 3 : 1;
  for (int p = 0; p < npairs; ++p) {
    red[tid] = pacc[p]; __syncthreads();
    for (int st = 128; st > 0; st >>= 1) { if (tid < st) red[tid] += red[tid + st]; __syncthreads(); }
    if (tid == 0) simPart[((g == 3) ? (3 + p) : g) * B_ + b] = red[0];
    __syncthreads();
  }
}

// =====================================================================
// K3: finalize sims (means) + L1 norms -> 8 scalar outputs (single block)
// order: sim_sf, sim_st, sim_ft, sim_sft, l1_st, l1_sf, l1_ft, l1_sft
// =====================================================================
__global__ __launch_bounds__(256) void finalize_kernel(
    const float* __restrict__ simPart,
    const float* __restrict__ Ssf, const float* __restrict__ Sst,
    const float* __restrict__ Sft, const float* __restrict__ Ssft,
    float* __restrict__ outScal) {
  __shared__ float red[256];
  int tid = threadIdx.x;
  float s[6];
  for (int slot = 0; slot < 6; ++slot) {
    red[tid] = simPart[slot * B_ + tid];
    __syncthreads();
    for (int st = 128; st > 0; st >>= 1) { if (tid < st) red[tid] += red[tid + st]; __syncthreads(); }
    s[slot] = red[0];
    __syncthreads();
  }
  const float inv = 1.f / (float)OUT1;
  if (tid == 0) {
    outScal[0] = s[0] * inv;
    outScal[1] = s[1] * inv;
    outScal[2] = s[2] * inv;
    outScal[3] = (s[3] + s[4] + s[5]) * inv;
  }
  const float* Sl[4] = {Sst, Ssf, Sft, Ssft};
  for (int q = 0; q < 4; ++q) {
    float acc = 0.f;
    for (int e = tid; e < NN; e += 256) acc += fabsf(Sl[q][e]);
    red[tid] = acc; __syncthreads();
    for (int st = 128; st > 0; st >>= 1) { if (tid < st) red[tid] += red[tid + st]; __syncthreads(); }
    if (tid == 0) outScal[4 + q] = red[0];
    __syncthreads();
  }
}

// =====================================================================
// K4: GRU.  Per (g,b): x1=[o,h] WMMA-> r,z ; x2=[r*h,o] WMMA-> hh ;
// out = (1-z)h + z*tanh.  K=512 (16 k-steps), M padded 35->48.
// =====================================================================
__global__ __launch_bounds__(256) void gru_kernel(
    const float* __restrict__ O, const float* __restrict__ com, P7 Hh,
    const _Float16* __restrict__ WrS, const _Float16* __restrict__ WzS,
    const _Float16* __restrict__ WhS, float* __restrict__ out) {
  int g = blockIdx.y, b = blockIdx.x;
  __shared__ __align__(32) _Float16 xsw[3 * 16 * 32 * 16];   // [mtile][kstep][lane][i] (48 KB)
  const float* op = (g < 3) ? (O + ((size_t)g * B_ + b) * NH)
                            : (com + ((size_t)(g - 3) * B_ + b) * NH);
  const float* hp = Hh.p[g] + (size_t)b * NH;
  // x1 = [o | h], padded rows zero, A-fragment swizzle
  for (int e = threadIdx.x; e < 24576; e += 256) {
    int i = e & 15, lane = (e >> 4) & 31, kstep = (e >> 9) & 15, mtile = e >> 13;
    int hl = lane >> 4;
    int m = mtile * 16 + (lane & 15);
    int k = kstep * 32 + ((i >> 3) << 4) + (hl << 3) + (i & 7);
    float v = 0.f;
    if (m < N_) v = (k < H_) ? op[m * H_ + k] : hp[m * H_ + (k - H_)];
    xsw[e] = (_Float16)v;
  }
  __syncthreads();
  int wave = threadIdx.x >> 5, lane = threadIdx.x & 31;
  int hl = lane >> 4;
  v8f accR[6], accZ[6];
  for (int t = 0; t < 6; ++t) {
    int tile = wave * 6 + t;
    int mtile = tile >> 4, ntile = tile & 15;
    v8f ar = {0.f,0.f,0.f,0.f,0.f,0.f,0.f,0.f};
    v8f az = {0.f,0.f,0.f,0.f,0.f,0.f,0.f,0.f};
    for (int ks = 0; ks < 16; ++ks) {
      v16h av  = *(const v16h*)&xsw[((mtile * 16 + ks) * 32 + lane) * 16];
      v16h brv = *(const v16h*)&WrS[((((size_t)g * 16 + ks) * 16 + ntile) * 32 + lane) * 16];
      v16h bzv = *(const v16h*)&WzS[((((size_t)g * 16 + ks) * 16 + ntile) * 32 + lane) * 16];
      ar = wmma_f16(av, brv, ar);
      az = wmma_f16(av, bzv, az);
    }
    accR[t] = ar; accZ[t] = az;
  }
  __syncthreads();           // everyone done reading x1 fragments
  // rewrite LDS with x2 = [r*h | o] (tiles partition (m,n) space exactly once)
  for (int t = 0; t < 6; ++t) {
    int tile = wave * 6 + t;
    int mtile = tile >> 4, ntile = tile & 15;
    int n = ntile * 16 + (lane & 15);
    for (int j = 0; j < 8; ++j) {
      int m = mtile * 16 + j + hl * 8;
      float rv = 1.f / (1.f + __expf(-accR[t][j]));
      float rh = (m < N_) ? rv * hp[m * H_ + n] : 0.f;
      float ov = (m < N_) ? op[m * H_ + n] : 0.f;
      for (int part = 0; part < 2; ++part) {
        int k = (part == 0) ? n : (H_ + n);
        float v = (part == 0) ? rh : ov;
        int kstep = k >> 5, kk = k & 31;
        int hk = (kk >> 3) & 1;
        int i  = ((kk >> 4) << 3) | (kk & 7);
        int lw = hk * 16 + (m & 15);
        xsw[(((m >> 4) * 16 + kstep) * 32 + lw) * 16 + i] = (_Float16)v;
      }
    }
  }
  __syncthreads();
  for (int t = 0; t < 6; ++t) {
    int tile = wave * 6 + t;
    int mtile = tile >> 4, ntile = tile & 15;
    v8f ah = {0.f,0.f,0.f,0.f,0.f,0.f,0.f,0.f};
    for (int ks = 0; ks < 16; ++ks) {
      v16h av  = *(const v16h*)&xsw[((mtile * 16 + ks) * 32 + lane) * 16];
      v16h bhv = *(const v16h*)&WhS[((((size_t)g * 16 + ks) * 16 + ntile) * 32 + lane) * 16];
      ah = wmma_f16(av, bhv, ah);
    }
    int n = ntile * 16 + (lane & 15);
    for (int j = 0; j < 8; ++j) {
      int m = mtile * 16 + j + hl * 8;
      if (m < N_) {
        float z  = 1.f / (1.f + __expf(-accZ[t][j]));
        float e2 = __expf(2.f * ah[j]);
        float hh = (e2 - 1.f) / (e2 + 1.f);
        float hv = hp[m * H_ + n];
        out[((size_t)g * B_ + b) * NH + m * H_ + n] = (1.f - z) * hv + z * hh;
      }
    }
  }
}

// =====================================================================
extern "C" void kernel_launch(void* const* d_in, const int* in_sizes, int n_in,
                              void* d_out, int out_size, void* d_ws, size_t ws_size,
                              hipStream_t stream) {
  (void)in_sizes; (void)n_in; (void)out_size; (void)ws_size;
  const float* As   = (const float*)d_in[12];
  const float* Af   = (const float*)d_in[13];
  const float* At   = (const float*)d_in[14];
  const float* V1   = (const float*)d_in[22];
  const float* c1   = (const float*)d_in[23];
  const float* V2   = (const float*)d_in[24];
  const float* c2   = (const float*)d_in[25];
  const float* Ssf  = (const float*)d_in[26];
  const float* Sst  = (const float*)d_in[27];
  const float* Sft  = (const float*)d_in[28];
  const float* Ssft = (const float*)d_in[29];
  const float* Wr   = (const float*)d_in[30];
  const float* Wz   = (const float*)d_in[31];
  const float* Wh   = (const float*)d_in[32];

  char* ws = (char*)d_ws;
  float*    A1   = (float*)(ws + OFF_A1);
  float*    A2   = (float*)(ws + OFF_A2);
  _Float16* W1s  = (_Float16*)(ws + OFF_W1);
  _Float16* W2s  = (_Float16*)(ws + OFF_W2);
  _Float16* WrS  = (_Float16*)(ws + OFF_WR);
  _Float16* WzS  = (_Float16*)(ws + OFF_WZ);
  _Float16* WhS  = (_Float16*)(ws + OFF_WH);
  float*    O    = (float*)(ws + OFF_O);
  float*    com  = (float*)(ws + OFF_COM);
  float*    simP = (float*)(ws + OFF_SIM);
  float*    outv = (float*)d_out;

  P12 X; for (int r = 0; r < 12; ++r) X.p[r] = (const float*)d_in[r];
  P7  Hh; for (int g = 0; g < 7; ++g) Hh.p[g] = (const float*)d_in[15 + g];

  compose_kernel<<<1, 256, 0, stream>>>(As, Af, At, A1, A2);
  gcnw_kernel<<<(2 * R_ * 16384) / 256, 256, 0, stream>>>(V1, c1, V2, c2, W1s, W2s);
  gatew_kernel<<<(3 * 7 * 131072) / 256, 256, 0, stream>>>(Wr, Wz, Wh, WrS, WzS, WhS);

  dim3 gGcn(B_, R_);
  gcn_kernel<<<gGcn, 256, 0, stream>>>(X, A1, A2, W1s, W2s, O);

  dim3 gMsh(B_, 4);
  msh_kernel<<<gMsh, 256, 0, stream>>>(O, Ssf, Sst, Sft, Ssft, com, simP);

  finalize_kernel<<<1, 256, 0, stream>>>(simP, Ssf, Sst, Sft, Ssft, outv + (size_t)7 * OUT1);

  dim3 gGru(B_, 7);
  gru_kernel<<<gGru, 256, 0, stream>>>(O, com, Hh, WrS, WzS, WhS, outv);
}
